// QuantizedCNNLSTMModelV2_62551903699533
// MI455X (gfx1250) — compile-verified, matrix operand order checked by reading, not run on hardware
//
#include <hip/hip_runtime.h>
#include <hip/hip_bf16.h>
#include <cstdint>

typedef __attribute__((ext_vector_type(16))) _Float16 v16h;
typedef __attribute__((ext_vector_type(8)))  float    v8f;
typedef __attribute__((ext_vector_type(4)))  unsigned int v4u;
typedef __attribute__((ext_vector_type(8)))  int      v8i;
typedef __attribute__((ext_vector_type(4)))  int      v4i;

// ---------------------------------------------------------------------------
// WMMA helpers (gfx1250, wave32). D = A(16xK) * B(Kx16) + C, K chunked by 32.
// 16-bit operand layout: lane l -> row/col = l&15, half = l>>4; the lane's 16
// elements are k = 8*half+0..7 and k = 16+8*half+0..7  (two contiguous 16B runs
// when K is contiguous in memory -> two b128 loads).
// C/D layout: lane l -> column n = l&15; vgpr v -> row m = v + 8*(l>>4).
// ---------------------------------------------------------------------------
__device__ inline v16h load_frag_kcontig(const _Float16* base_k, int half) {
  union { uint4 u[2]; v16h h; } r;
  r.u[0] = *(const uint4*)(base_k + half * 8);
  r.u[1] = *(const uint4*)(base_k + 16 + half * 8);
  return r.h;
}

template <typename F>
__device__ inline v16h make_frag_b(int lane, F f) {  // gather path (convs)
  v16h r;
  const int n = lane & 15, half = lane >> 4;
#pragma unroll
  for (int v = 0; v < 8; ++v) {
    const int kb = (v < 4) ? (v * 2 + half * 8) : (16 + (v - 4) * 2 + half * 8);
    r[2 * v]     = f(kb, n);
    r[2 * v + 1] = f(kb + 1, n);
  }
  return r;
}

__device__ inline v8f wmma_f16f32(v16h a, v16h b, v8f c) {
  return __builtin_amdgcn_wmma_f32_16x16x32_f16(false, a, false, b, (short)0, c,
                                                false, false);
}

__device__ inline float sigm(float x) { return 1.0f / (1.0f + __expf(-x)); }

// ---------------------------------------------------------------------------
// CDNA5 async global->LDS copy (ASYNCcnt-tracked). Generic shared pointers
// carry the LDS byte offset in their low 32 bits (ISA 10.2 aperture rules).
// ---------------------------------------------------------------------------
__device__ inline void async_g2l_b128(void* lds_ptr, const void* gptr) {
  const unsigned loff = (unsigned)(uintptr_t)lds_ptr;
  asm volatile("global_load_async_to_lds_b128 %0, %1, off"
               :: "v"(loff), "v"(gptr) : "memory");
}
__device__ inline void wait_async0() {
  asm volatile("s_wait_asynccnt 0x0" ::: "memory");
}

// ---------------------------------------------------------------------------
// CDNA5 Tensor Data Mover: stage a 512x128 f16 (128KB) matrix into LDS with a
// single D#-descriptor DMA (TENSORcnt-tracked). D# per ISA 8.3-8.5:
//   group0: count=1 | lds_addr | global_addr[56:0] | type=2
//   group1: data_size=2B, tensor 128x512 (dim0 stride 128), tile = tensor
// Called by ONE wave (TDM ignores EXEC; instruction issues once per wave).
// Falls back to ASYNCcnt copies if the builtin is unavailable.
// ---------------------------------------------------------------------------
__device__ inline void tdm_load_512x128_f16(void* lds_dst, const void* gsrc,
                                            int lane) {
#if __has_builtin(__builtin_amdgcn_tensor_load_to_lds)
  const unsigned la = (unsigned)(uintptr_t)lds_dst;
  const unsigned long long ga = (unsigned long long)(uintptr_t)gsrc;
  v4u g0;
  g0[0] = 1u;                                    // count=1 (valid), user D#
  g0[1] = la;                                    // lds_addr (bytes)
  g0[2] = (unsigned)(ga & 0xFFFFFFFFu);          // global_addr[31:0]
  g0[3] = (unsigned)((ga >> 32) & 0x01FFFFFFu)   // global_addr[56:32]
          | (2u << 30);                          // type=2 ("image")
  v8i g1;
  g1[0] = (1 << 16);    // workgroup_mask=0 (no cluster), data_size=1 -> 2B
  g1[1] = (128 << 16);  // tensor_dim0 = 128 (low 16 bits)
  g1[2] = (512 << 16);  // tensor_dim1 = 512 (low 16 bits)
  g1[3] = (128 << 16);  // tile_dim0 = 128
  g1[4] = 512;          // tile_dim1 = 512, tile_dim2 = 0 (unused)
  g1[5] = 128;          // tensor_dim0_stride = 128
  g1[6] = 0;
  g1[7] = 0;
  const v4i z4 = {0, 0, 0, 0};
#if __clang_major__ >= 23
  const v8i z8 = {0, 0, 0, 0, 0, 0, 0, 0};
  __builtin_amdgcn_tensor_load_to_lds(g0, g1, z4, z4, z8, 0);
#else
  __builtin_amdgcn_tensor_load_to_lds(g0, g1, z4, z4, 0);
#endif
#if __has_builtin(__builtin_amdgcn_s_wait_tensorcnt)
  __builtin_amdgcn_s_wait_tensorcnt((short)0);
#else
  asm volatile("s_wait_tensorcnt 0x0" ::: "memory");
#endif
#else
  for (int i = lane; i < 8192; i += 32)
    async_g2l_b128((char*)lds_dst + i * 16, (const char*)gsrc + i * 16);
  wait_async0();
#endif
}

// ---------------------------------------------------------------------------
// 1) Per-tensor max|w| (atomicMax on non-negative float bit pattern)
// ---------------------------------------------------------------------------
__global__ void maxabs_kernel(const float* __restrict__ w, int n,
                              unsigned int* __restrict__ out) {
  float m = 0.0f;
  for (int i = blockIdx.x * blockDim.x + threadIdx.x; i < n;
       i += gridDim.x * blockDim.x)
    m = fmaxf(m, fabsf(w[i]));
#pragma unroll
  for (int off = 16; off > 0; off >>= 1)
    m = fmaxf(m, __shfl_down(m, off, 32));
  if ((threadIdx.x & 31) == 0) atomicMax(out, __float_as_uint(m));
}

// 2) q4 fake-quant (s = max|w|/7, round/clip to [-8,7]) -> f16
__global__ void quant_q4_f16(const float* __restrict__ w, int n,
                             const unsigned int* __restrict__ maxbits,
                             _Float16* __restrict__ out) {
  const int i = blockIdx.x * blockDim.x + threadIdx.x;
  if (i >= n) return;
  const float s = fmaxf(__uint_as_float(*maxbits) / 7.0f, 1e-8f);
  const float q = fminf(fmaxf(rintf(w[i] / s), -8.0f), 7.0f) * s;
  out[i] = (_Float16)q;
}

// 2b) same, re-packing rows [COUT,KTOT] -> [COUT,KPAD] zero-padded so conv
//     A-fragments can use aligned b128 loads with no bounds checks.
__global__ void quant_q4_f16_pad(const float* __restrict__ w, int ktot,
                                 int kpad, int total,
                                 const unsigned int* __restrict__ maxbits,
                                 _Float16* __restrict__ out) {
  const int i = blockIdx.x * blockDim.x + threadIdx.x;
  if (i >= total) return;
  const int co = i / kpad, kk = i % kpad;
  float q = 0.0f;
  if (kk < ktot) {
    const float s = fmaxf(__uint_as_float(*maxbits) / 7.0f, 1e-8f);
    q = fminf(fmaxf(rintf(w[co * ktot + kk] / s), -8.0f), 7.0f) * s;
  }
  out[i] = (_Float16)q;
}

// ---------------------------------------------------------------------------
// 3) Implicit-GEMM stride-4 SAME 3x3 conv + bias + relu via WMMA.
// ---------------------------------------------------------------------------
template <int CIN, int COUT, int HIN, int WIN, int HOUT, int WOUT, int PADH,
          int PADW, typename TIN>
__global__ void conv_s4_wmma(const TIN* __restrict__ in,
                             const _Float16* __restrict__ wq,
                             const float* __restrict__ bias,
                             _Float16* __restrict__ out, int BS) {
  constexpr int KTOT = CIN * 9;
  constexpr int KCH  = (KTOT + 31) / 32;
  constexpr int KPAD = KCH * 32;
  const int P    = BS * HOUT * WOUT;
  const int lane = threadIdx.x & 31;
  const int half = lane >> 4;
  const int wave = threadIdx.x >> 5;
  const int base = (blockIdx.x * 4 + wave) * 16;
  if (base >= P) return;
  const int coutBase = blockIdx.y * 16;

  const int mRow = min(coutBase + (lane & 15), COUT - 1);
  const _Float16* arow = wq + mRow * KPAD;

  const int  n     = lane & 15;
  const int  pos   = base + n;
  const bool store = pos < P;
  const int  pp    = store ? pos : P - 1;
  const int  bs    = pp / (HOUT * WOUT);
  const int  rem   = pp % (HOUT * WOUT);
  const int  oh    = rem / WOUT;
  const int  ow    = rem % WOUT;
  const TIN* inb   = in + (size_t)bs * CIN * HIN * WIN;

  __builtin_prefetch(inb + (oh * 4 * WIN + ow * 4), 0, 1);  // global_prefetch

  v8f acc = {};
#pragma unroll
  for (int kc = 0; kc < KCH; ++kc) {
    const int kb0 = kc * 32;
    v16h a = load_frag_kcontig(arow + kb0, half);
    v16h b = make_frag_b(lane, [&](int k, int) -> _Float16 {
      const int kk = kb0 + k;
      if (kk >= KTOT) return (_Float16)0.0f;
      const int c  = kk / 9;
      const int r9 = kk % 9;
      const int ih = oh * 4 - PADH + (r9 / 3);
      const int iw = ow * 4 - PADW + (r9 % 3);
      if constexpr (PADH != 0 || PADW != 0) {
        if (ih < 0 || ih >= HIN || iw < 0 || iw >= WIN) return (_Float16)0.0f;
      }
      return (_Float16)(float)inb[(c * HIN + ih) * WIN + iw];
    });
    acc = wmma_f16f32(a, b, acc);
  }

  if (store) {
#pragma unroll
    for (int v = 0; v < 8; ++v) {
      const int m  = v + 8 * half;
      const int co = coutBase + m;
      if (co < COUT) {
        float val = acc[v] + bias[co];
        val       = val > 0.0f ? val : 0.0f;
        out[((size_t)bs * COUT + co) * (HOUT * WOUT) + oh * WOUT + ow] =
            (_Float16)val;
      }
    }
  }
}

// ---------------------------------------------------------------------------
// 4) Global average pool [BS,32,7,10] -> f16 [BS,32]
// ---------------------------------------------------------------------------
__global__ void pool_kernel(const _Float16* __restrict__ in,
                            _Float16* __restrict__ out, int total) {
  const int idx = blockIdx.x * blockDim.x + threadIdx.x;
  if (idx >= total) return;
  const _Float16* p = in + (size_t)idx * 70;
  float s = 0.0f;
#pragma unroll
  for (int i = 0; i < 70; ++i) s += (float)p[i];
  out[idx] = (_Float16)(s * (1.0f / 70.0f));
}

// ---------------------------------------------------------------------------
// 5) feat linear: relu(pooled[BS,32] @ W[128,32]^T + b) via WMMA (K=32)
// ---------------------------------------------------------------------------
__global__ void feat_wmma(const _Float16* __restrict__ pooled,
                          const _Float16* __restrict__ wq,
                          const float* __restrict__ bias,
                          _Float16* __restrict__ feat_h, int BS) {
  const int lane = threadIdx.x & 31;
  const int half = lane >> 4;
  const int mBase = blockIdx.y * 16;
  const int col   = blockIdx.x * 16 + (lane & 15);
  const int colc  = col < BS ? col : BS - 1;

  v16h a = load_frag_kcontig(wq + (mBase + (lane & 15)) * 32, half);
  v16h b = load_frag_kcontig(pooled + colc * 32, half);
  v8f acc = {};
  acc = wmma_f16f32(a, b, acc);

  if (col < BS) {
#pragma unroll
    for (int v = 0; v < 8; ++v) {
      const int m = mBase + v + 8 * half;
      float val   = acc[v] + bias[m];
      val         = val > 0.0f ? val : 0.0f;
      feat_h[col * 128 + m] = (_Float16)val;
    }
  }
}

// ---------------------------------------------------------------------------
// 6) Hoisted time-parallel x@Wih^T GEMM (biases folded): xw[n][512].
// ---------------------------------------------------------------------------
__global__ void gemm_xw_wmma(const _Float16* __restrict__ xin,  // [N,K]
                             const _Float16* __restrict__ w,    // [512,K]
                             const float* __restrict__ b0,
                             const float* __restrict__ b1,
                             float* __restrict__ xw, int N, int K) {
  const int lane = threadIdx.x & 31;
  const int half = lane >> 4;
  const int mBase = blockIdx.y * 16;
  const int col   = blockIdx.x * 16 + (lane & 15);
  const int colc  = col < N ? col : N - 1;
  const _Float16* arow = w + (size_t)(mBase + (lane & 15)) * K;
  const _Float16* brow = xin + (size_t)colc * K;

  v8f acc = {};
  for (int kb = 0; kb < K; kb += 32)
    acc = wmma_f16f32(load_frag_kcontig(arow + kb, half),
                      load_frag_kcontig(brow + kb, half), acc);

  if (col < N) {
#pragma unroll
    for (int v = 0; v < 8; ++v) {
      const int m = mBase + v + 8 * half;
      xw[(size_t)col * 512 + m] = acc[v] + b0[m] + b1[m];
    }
  }
}

// ---------------------------------------------------------------------------
// 7) Decoder input GEMM, K=6400 split across 25 blocks of 256, f32 atomics.
// ---------------------------------------------------------------------------
__global__ void dec_xw_wmma(const _Float16* __restrict__ xin,  // [2,6400]
                            const _Float16* __restrict__ wih,  // [512,6400]
                            float* __restrict__ xw) {          // [2,512] zeroed
  const int lane = threadIdx.x & 31;
  const int half = lane >> 4;
  const int mBase  = blockIdx.y * 16;
  const int kStart = blockIdx.x * 256;
  const int n      = lane & 15;
  const _Float16* arow = wih + (size_t)(mBase + n) * 6400;
  const _Float16* brow = xin + (size_t)(n < 2 ? n : 1) * 6400;

  v8f acc = {};
#pragma unroll
  for (int kc = 0; kc < 8; ++kc) {
    const int kb = kStart + kc * 32;
    acc = wmma_f16f32(load_frag_kcontig(arow + kb, half),
                      load_frag_kcontig(brow + kb, half), acc);
  }
  if (n < 2) {
#pragma unroll
    for (int v = 0; v < 8; ++v)
      atomicAdd(&xw[n * 512 + mBase + v + 8 * half], acc[v]);
  }
}

// ---------------------------------------------------------------------------
// 8) Persistent LSTM recurrence. Whh (128KB f16) staged into LDS once:
//    encoder via TDM tensor_load_to_lds, decoder via async-LDS b128 copies
//    (both CDNA5 data-mover paths). Each step: 4 WMMA/wave on LDS operands.
// ---------------------------------------------------------------------------
template <int TSTEPS, bool XW_PER_STEP, bool USE_TDM>
__global__ __launch_bounds__(1024) void lstm_rec_wmma(
    const float* __restrict__ xw,       // enc: [100,512] (+bias); dec: [2,512]
    const _Float16* __restrict__ whh,   // [512,128] f16 q4
    const float* __restrict__ bih,      // dec only
    const float* __restrict__ bhh,
    _Float16* __restrict__ hseq16,      // enc: [2,50,128] f16 out
    float* __restrict__ hseq32) {       // dec: [2,5,128] f32 out
  __shared__ __align__(16) _Float16 whh_l[512 * 128];  // 128KB
  __shared__ __align__(16) _Float16 hbuf[256];
  __shared__ float cbuf[256];
  __shared__ float gates[1024];
  const int tid  = threadIdx.x;
  const int lane = tid & 31;
  const int half = lane >> 4;
  const int mBase = (tid >> 5) * 16;
  const int n     = lane & 15;

  if constexpr (USE_TDM) {
    if (tid < 32) tdm_load_512x128_f16(whh_l, whh, lane);  // one TDM op, wave 0
  } else {
#pragma unroll
    for (int i = 0; i < 8; ++i) {
      const int off = (tid + i * 1024) * 16;
      async_g2l_b128((char*)whh_l + off, (const char*)whh + off);
    }
    wait_async0();
  }
  if (tid < 256) { hbuf[tid] = (_Float16)0.0f; cbuf[tid] = 0.0f; }
  __syncthreads();

  const _Float16* arow = whh_l + (mBase + n) * 128;
  const _Float16* bcol = hbuf + (n < 2 ? n : 1) * 128;

  for (int t = 0; t < TSTEPS; ++t) {
    v8f acc = {};
#pragma unroll
    for (int kc = 0; kc < 4; ++kc)
      acc = wmma_f16f32(load_frag_kcontig(arow + kc * 32, half),
                        load_frag_kcontig(bcol + kc * 32, half), acc);
    if (n < 2) {
#pragma unroll
      for (int v = 0; v < 8; ++v) {
        const int m = mBase + v + 8 * half;
        float g = acc[v];
        if (XW_PER_STEP) g += xw[(size_t)(n * TSTEPS + t) * 512 + m];
        else             g += xw[n * 512 + m] + bih[m] + bhh[m];
        gates[n * 512 + m] = g;
      }
    }
    __syncthreads();
    if (tid < 256) {
      const int b = tid >> 7, j = tid & 127;
      const float gi = sigm(gates[b * 512 + j]);
      const float gf = sigm(gates[b * 512 + 128 + j]);
      const float gg = tanhf(gates[b * 512 + 256 + j]);
      const float go = sigm(gates[b * 512 + 384 + j]);
      const float c  = gf * cbuf[tid] + gi * gg;
      const float h  = go * tanhf(c);
      cbuf[tid] = c;
      hbuf[tid] = (_Float16)h;
      if (XW_PER_STEP) hseq16[(b * TSTEPS + t) * 128 + j] = (_Float16)h;
      else             hseq32[(b * TSTEPS + t) * 128 + j] = h;
    }
    __syncthreads();
  }
}

// 9) fc head: out[b,t,o] = dec_h[b,t,:] . fc_w[o,:] + fc_b[o]  (30 outputs)
__global__ void fc_kernel(const float* __restrict__ hin,
                          const _Float16* __restrict__ wq,
                          const float* __restrict__ fb,
                          float* __restrict__ out) {
  const int idx = blockIdx.x * blockDim.x + threadIdx.x;
  if (idx >= 30) return;
  const int bt = idx / 3, o = idx % 3;
  float s = fb[o];
#pragma unroll 16
  for (int j = 0; j < 128; ++j) s += hin[bt * 128 + j] * (float)wq[o * 128 + j];
  out[idx] = s;
}

// ---------------------------------------------------------------------------
// Host orchestration
// ---------------------------------------------------------------------------
extern "C" void kernel_launch(void* const* d_in, const int* in_sizes, int n_in,
                              void* d_out, int out_size, void* d_ws,
                              size_t ws_size, hipStream_t stream) {
  (void)in_sizes; (void)n_in; (void)out_size; (void)ws_size;
  const float* x       = (const float*)d_in[0];
  const float* conv1_w = (const float*)d_in[1];
  const float* conv1_b = (const float*)d_in[2];
  const float* conv2_w = (const float*)d_in[3];
  const float* conv2_b = (const float*)d_in[4];
  const float* conv3_w = (const float*)d_in[5];
  const float* conv3_b = (const float*)d_in[6];
  const float* feat_w  = (const float*)d_in[7];
  const float* feat_b  = (const float*)d_in[8];
  const float* enc_wih = (const float*)d_in[9];
  const float* enc_whh = (const float*)d_in[10];
  const float* enc_bih = (const float*)d_in[11];
  const float* enc_bhh = (const float*)d_in[12];
  const float* dec_wih = (const float*)d_in[13];
  const float* dec_whh = (const float*)d_in[14];
  const float* dec_bih = (const float*)d_in[15];
  const float* dec_bhh = (const float*)d_in[16];
  const float* fc_w    = (const float*)d_in[17];
  const float* fc_b    = (const float*)d_in[18];
  float* out           = (float*)d_out;

  char*  ws = (char*)d_ws;
  size_t o  = 0;
  auto alloc = [&](size_t bytes) -> void* {
    void* p = ws + o;
    o       = (o + bytes + 255) & ~(size_t)255;
    return p;
  };
  unsigned int* maxabs = (unsigned int*)alloc(16 * sizeof(unsigned int));
  float*    xw_dec     = (float*)alloc(1024 * sizeof(float));
  const size_t zeroBytes = o;  // maxabs + xw_dec zeroed each call
  _Float16* conv1_q   = (_Float16*)alloc((size_t)8 * 32 * 2);    // KPAD=32
  _Float16* conv2_q   = (_Float16*)alloc((size_t)16 * 96 * 2);   // KPAD=96
  _Float16* conv3_q   = (_Float16*)alloc((size_t)32 * 160 * 2);  // KPAD=160
  _Float16* feat_q    = (_Float16*)alloc(4096 * 2);
  _Float16* enc_wih_q = (_Float16*)alloc(65536 * 2);
  _Float16* enc_whh_q = (_Float16*)alloc(65536 * 2);
  _Float16* dec_wih_q = (_Float16*)alloc((size_t)3276800 * 2);
  _Float16* dec_whh_q = (_Float16*)alloc(65536 * 2);
  _Float16* fc_q      = (_Float16*)alloc(384 * 2);
  _Float16* act1      = (_Float16*)alloc((size_t)100 * 8 * 100 * 160 * 2);
  _Float16* act2      = (_Float16*)alloc((size_t)100 * 16 * 25 * 40 * 2);
  _Float16* act3      = (_Float16*)alloc((size_t)100 * 32 * 70 * 2);
  _Float16* pooled    = (_Float16*)alloc(3200 * 2);
  _Float16* feat_h    = (_Float16*)alloc(12800 * 2);
  _Float16* enc_out   = (_Float16*)alloc(12800 * 2);
  float*    xw_enc    = (float*)alloc((size_t)100 * 512 * sizeof(float));
  float*    dec_h     = (float*)alloc(1280 * sizeof(float));

  hipMemsetAsync(d_ws, 0, zeroBytes, stream);

  auto cdiv = [](long a, long b) -> unsigned { return (unsigned)((a + b - 1) / b); };

  // max|w| for all 9 quantized tensors
  struct MW { const float* w; int n; int slot; };
  const MW mws[9] = {{conv1_w, 216, 0},     {conv2_w, 1152, 1},
                     {conv3_w, 4608, 2},    {feat_w, 4096, 3},
                     {enc_wih, 65536, 4},   {enc_whh, 65536, 5},
                     {dec_wih, 3276800, 6}, {dec_whh, 65536, 7},
                     {fc_w, 384, 8}};
  for (int i = 0; i < 9; ++i) {
    unsigned blocks = cdiv(mws[i].n, 256);
    if (blocks > 1024) blocks = 1024;
    maxabs_kernel<<<blocks, 256, 0, stream>>>(mws[i].w, mws[i].n,
                                              &maxabs[mws[i].slot]);
  }
  // conv weights: quantize into K-padded rows for b128 A-fragments
  quant_q4_f16_pad<<<cdiv(8 * 32, 256), 256, 0, stream>>>(
      conv1_w, 27, 32, 8 * 32, &maxabs[0], conv1_q);
  quant_q4_f16_pad<<<cdiv(16 * 96, 256), 256, 0, stream>>>(
      conv2_w, 72, 96, 16 * 96, &maxabs[1], conv2_q);
  quant_q4_f16_pad<<<cdiv(32 * 160, 256), 256, 0, stream>>>(
      conv3_w, 144, 160, 32 * 160, &maxabs[2], conv3_q);
  // dense weights: plain quantize -> f16
  struct QW { const float* w; int n; int slot; _Float16* dst; };
  const QW qws[6] = {{feat_w, 4096, 3, feat_q},
                     {enc_wih, 65536, 4, enc_wih_q},
                     {enc_whh, 65536, 5, enc_whh_q},
                     {dec_wih, 3276800, 6, dec_wih_q},
                     {dec_whh, 65536, 7, dec_whh_q},
                     {fc_w, 384, 8, fc_q}};
  for (int i = 0; i < 6; ++i)
    quant_q4_f16<<<cdiv(qws[i].n, 256), 256, 0, stream>>>(
        qws[i].w, qws[i].n, &maxabs[qws[i].slot], qws[i].dst);

  // CNN backbone (implicit-GEMM WMMA convs)
  const int BS = 100;
  {
    const long P = (long)BS * 100 * 160;
    conv_s4_wmma<3, 8, 400, 640, 100, 160, 0, 0, float>
        <<<dim3(cdiv(P, 64), 1), 128, 0, stream>>>(x, conv1_q, conv1_b, act1, BS);
  }
  {
    const long P = (long)BS * 25 * 40;
    conv_s4_wmma<8, 16, 100, 160, 25, 40, 0, 0, _Float16>
        <<<dim3(cdiv(P, 64), 1), 128, 0, stream>>>(act1, conv2_q, conv2_b, act2, BS);
  }
  {
    const long P = (long)BS * 7 * 10;
    conv_s4_wmma<16, 32, 25, 40, 7, 10, 1, 0, _Float16>
        <<<dim3(cdiv(P, 64), 2), 128, 0, stream>>>(act2, conv3_q, conv3_b, act3, BS);
  }

  // pool + feat
  pool_kernel<<<cdiv(3200, 256), 256, 0, stream>>>(act3, pooled, 3200);
  feat_wmma<<<dim3(cdiv(BS, 16), 8), 32, 0, stream>>>(pooled, feat_q, feat_b,
                                                      feat_h, BS);

  // encoder: hoisted time-parallel x@Wih (bias folded), then TDM-staged recurrence
  gemm_xw_wmma<<<dim3(cdiv(100, 16), 32), 32, 0, stream>>>(
      feat_h, enc_wih_q, enc_bih, enc_bhh, xw_enc, 100, 128);
  lstm_rec_wmma<50, true, true><<<1, 1024, 0, stream>>>(
      xw_enc, enc_whh_q, enc_bih, enc_bhh, enc_out, (float*)nullptr);

  // decoder: hoisted x@Wih (input identical across 5 steps), async-staged recurrence
  dec_xw_wmma<<<dim3(25, 32), 32, 0, stream>>>(enc_out, dec_wih_q, xw_dec);
  lstm_rec_wmma<5, false, false><<<1, 1024, 0, stream>>>(
      xw_dec, dec_whh_q, dec_bih, dec_bhh, (_Float16*)nullptr, dec_h);

  // head
  fc_kernel<<<1, 64, 0, stream>>>(dec_h, fc_q, fc_b, out);
}
